// AdditiveAttention_61168924230076
// MI455X (gfx1250) — matrix operand order repaired and checked by reading, not compiled, not run on hardware
//
#include <hip/hip_runtime.h>
#include <hip/hip_bf16.h>
#include <math.h>

#define B_SZ 64
#define T_SZ 2048
#define E_SZ 256
#define I_SZ 256

typedef __attribute__((ext_vector_type(16))) __bf16 v16bf;
typedef __attribute__((ext_vector_type(8)))  float  v8f;

// Packed f32x2 -> bf16x2 in one dword via native v_cvt_pk_bf16_f32 (RNE).
// clang-22 does not declare the builtin for gfx1250, so emit it directly.
// Non-volatile asm: pure value computation, scheduler may reorder/CSE it.
__device__ __forceinline__ unsigned int f32x2_bf16x2(float a, float b) {
    unsigned int r;
    asm("v_cvt_pk_bf16_f32 %0, %1, %2" : "=v"(r) : "v"(a), "v"(b));
    return r;
}

// Branch-free tanh: native v_tanh_f32 on gfx1250 if the builtin exists,
// else exp2/rcp sequence (no EXEC-mask churn, no libm slow path).
__device__ __forceinline__ float fast_tanh(float x) {
#if defined(__has_builtin) && __has_builtin(__builtin_amdgcn_tanhf)
    return __builtin_amdgcn_tanhf(x);
#else
    float e = __builtin_amdgcn_exp2f(x * 2.8853900817779268f);
    return 1.0f - 2.0f * __builtin_amdgcn_rcpf(e + 1.0f);
#endif
}

__device__ __forceinline__ float fast_exp(float x) {
    return __builtin_amdgcn_exp2f(x * 1.4426950408889634f);   // v_exp_f32
}

union FragU {
    v16bf          v;
    unsigned int   u[8];
    uint4          q[2];
};

// ---------------------------------------------------------------------------
// Kernel 1: dec_proj[b][i] = sum_e w2[i][e] * dec[b][e]      (tiny, VALU fp32)
// ---------------------------------------------------------------------------
__global__ __launch_bounds__(256) void aa_decproj_kernel(
        const float* __restrict__ dec, const float* __restrict__ w2,
        float* __restrict__ dp) {
    const int b = blockIdx.x, i = threadIdx.x;
    const float4* wr = reinterpret_cast<const float4*>(w2 + (size_t)i * E_SZ);
    const float4* dr = reinterpret_cast<const float4*>(dec + (size_t)b * E_SZ);
    float acc = 0.f;
    #pragma unroll 8
    for (int e4 = 0; e4 < E_SZ / 4; ++e4) {
        float4 w = wr[e4], d = dr[e4];
        acc += w.x * d.x + w.y * d.y + w.z * d.z + w.w * d.w;
    }
    dp[(size_t)b * I_SZ + i] = acc;
}

// ---------------------------------------------------------------------------
// Kernel 2: scores[b][t] = v . tanh(W1 * enc[b][t] + dec_proj[b])
// bf16 WMMA GEMM fused with tanh + dot(v). One wave per 16-row t-strip.
// W1 staged to LDS as bf16 in two 128-row halves (64 KB static LDS).
// B fragments for a whole I-tile are prefetched into registers before the
// 8 back-to-back WMMAs so DS latency does not serialize the matrix pipe.
// ---------------------------------------------------------------------------
__global__ __launch_bounds__(256) void aa_scores_kernel(
        const float* __restrict__ enc, const float* __restrict__ w1,
        const float* __restrict__ vvec, const float* __restrict__ dp,
        float* __restrict__ scores) {
    __shared__ unsigned short w1lds[128 * E_SZ];   // 64 KB: half of W1 as bf16

    const int tid    = threadIdx.x;
    const int b      = blockIdx.y;
    const int wave   = tid >> 5;
    const int lane   = tid & 31;
    const int half   = lane >> 4;       // 0: lanes 0-15, 1: lanes 16-31
    const int lane16 = lane & 15;
    const int t0     = (blockIdx.x * 8 + wave) * 16;

    // --- Preload A fragments: row (t0+lane16) of encoder_states, K = 0..255.
    // ISA 16-bit A 16x32 layout: lanes 0-15 hold K {0..7,16..23},
    // lanes 16-31 hold K {8..15,24..31} (contiguous 8-element chunks).
    const float* arow = enc + ((size_t)b * T_SZ + t0 + lane16) * E_SZ;
    FragU afrag[8];
    #pragma unroll
    for (int kb = 0; kb < 8; ++kb) {
        const int e0 = kb * 32 + half * 8;
        #pragma unroll
        for (int c = 0; c < 2; ++c) {
            float4 f0 = *reinterpret_cast<const float4*>(arow + e0 + c * 16);
            float4 f1 = *reinterpret_cast<const float4*>(arow + e0 + c * 16 + 4);
            const int w0 = c * 4;                         // dword index in frag
            afrag[kb].u[w0 + 0] = f32x2_bf16x2(f0.x, f0.y);
            afrag[kb].u[w0 + 1] = f32x2_bf16x2(f0.z, f0.w);
            afrag[kb].u[w0 + 2] = f32x2_bf16x2(f1.x, f1.y);
            afrag[kb].u[w0 + 3] = f32x2_bf16x2(f1.z, f1.w);
        }
    }

    float sacc[8];
    #pragma unroll
    for (int r = 0; r < 8; ++r) sacc[r] = 0.f;

    for (int pass = 0; pass < 2; ++pass) {
        // Stage 128 rows of W1 -> LDS as bf16 (packed dword stores)
        for (int idx = tid * 4; idx < 128 * E_SZ; idx += 256 * 4) {
            float4 f = *reinterpret_cast<const float4*>(w1 + (size_t)pass * 128 * E_SZ + idx);
            uint2 p;
            p.x = f32x2_bf16x2(f.x, f.y);
            p.y = f32x2_bf16x2(f.z, f.w);
            *reinterpret_cast<uint2*>(&w1lds[idx]) = p;
        }
        __syncthreads();

        for (int ti = 0; ti < 8; ++ti) {
            const int irow_l = ti * 16 + lane16;              // row within LDS half
            const int irow_g = pass * 128 + irow_l;           // global i index
            const unsigned short* bbase = &w1lds[(size_t)irow_l * E_SZ];

            const float dpv = dp[(size_t)b * I_SZ + irow_g];
            const float vv  = vvec[irow_g];

            // Prefetch all B fragments for this I-tile (16x ds_load_b128).
            // B 32x16 bf16 layout: lanes 0-15 hold K {0..7,16..23} of column N,
            // lanes 16-31 hold K {8..15,24..31} (matches A).
            FragU bfr[8];
            #pragma unroll
            for (int kb = 0; kb < 8; ++kb) {
                const int k0 = kb * 32 + half * 8;
                bfr[kb].q[0] = *reinterpret_cast<const uint4*>(bbase + k0);
                bfr[kb].q[1] = *reinterpret_cast<const uint4*>(bbase + k0 + 16);
            }

            // 8 back-to-back WMMAs accumulate K = 256
            v8f c = {};
            #pragma unroll
            for (int kb = 0; kb < 8; ++kb)
                c = __builtin_amdgcn_wmma_f32_16x16x32_bf16(
                        false, afrag[kb].v, false, bfr[kb].v, (short)0, c, false, false);

            #pragma unroll
            for (int r = 0; r < 8; ++r)
                sacc[r] += fast_tanh(c[r] + dpv) * vv;
        }
        __syncthreads();   // before restaging LDS
    }

    // Reduce over the N dimension: 16 lanes in each half hold partial sums
    // (lanes 0-15 -> rows M=0..7, lanes 16-31 -> rows M=8..15).
    #pragma unroll
    for (int m = 1; m < 16; m <<= 1)
        #pragma unroll
        for (int r = 0; r < 8; ++r)
            sacc[r] += __shfl_xor(sacc[r], m, 32);

    if (lane16 == 0) {
        float* srow = scores + (size_t)b * T_SZ + t0 + half * 8;
        #pragma unroll
        for (int r = 0; r < 8; ++r) srow[r] = sacc[r];
    }
}

// ---------------------------------------------------------------------------
// Kernel 3: in-place softmax over T per batch row
// ---------------------------------------------------------------------------
__global__ __launch_bounds__(256) void aa_softmax_kernel(float* __restrict__ probs) {
    __shared__ float red[256];
    const int b = blockIdx.x, tid = threadIdx.x;
    float* row = probs + (size_t)b * T_SZ;

    float v[8];
    float m = -INFINITY;
    #pragma unroll
    for (int j = 0; j < 8; ++j) {
        v[j] = row[tid + j * 256];
        m = fmaxf(m, v[j]);
    }
    red[tid] = m; __syncthreads();
    for (int s = 128; s > 0; s >>= 1) {
        if (tid < s) red[tid] = fmaxf(red[tid], red[tid + s]);
        __syncthreads();
    }
    m = red[0]; __syncthreads();

    float sum = 0.f;
    #pragma unroll
    for (int j = 0; j < 8; ++j) { v[j] = fast_exp(v[j] - m); sum += v[j]; }
    red[tid] = sum; __syncthreads();
    for (int s = 128; s > 0; s >>= 1) {
        if (tid < s) red[tid] += red[tid + s];
        __syncthreads();
    }
    const float inv = 1.0f / red[0];
    #pragma unroll
    for (int j = 0; j < 8; ++j) row[tid + j * 256] = v[j] * inv;
}

// ---------------------------------------------------------------------------
// Kernel 4: ctx[b][e] = sum_t probs[b][t] * enc[b][t][e]  (2nd streaming pass)
// ---------------------------------------------------------------------------
__global__ __launch_bounds__(256) void aa_context_kernel(
        const float* __restrict__ enc, const float* __restrict__ probs,
        float* __restrict__ ctx) {
    __shared__ float p[T_SZ];
    const int b = blockIdx.x, e = threadIdx.x;
    #pragma unroll
    for (int j = 0; j < 8; ++j)
        p[e + j * 256] = probs[(size_t)b * T_SZ + e + j * 256];
    __syncthreads();

    const float* base = enc + (size_t)b * T_SZ * E_SZ + e;
    float acc = 0.f;
    #pragma unroll 4
    for (int t = 0; t < T_SZ; ++t)
        acc += p[t] * base[(size_t)t * E_SZ];
    ctx[(size_t)b * E_SZ + e] = acc;
}

// ---------------------------------------------------------------------------
extern "C" void kernel_launch(void* const* d_in, const int* in_sizes, int n_in,
                              void* d_out, int out_size, void* d_ws, size_t ws_size,
                              hipStream_t stream) {
    const float* enc = (const float*)d_in[0];   // [64,2048,256]
    const float* dec = (const float*)d_in[1];   // [64,256]
    const float* w1  = (const float*)d_in[2];   // [256,256]
    const float* w2  = (const float*)d_in[3];   // [256,256]
    const float* v   = (const float*)d_in[4];   // [1,256]

    float* out   = (float*)d_out;
    float* ctx   = out;                         // [64,256]
    float* probs = out + B_SZ * E_SZ;           // [64,2048] (scores -> probs in place)
    float* dp    = (float*)d_ws;                // [64,256] dec_proj scratch

    aa_decproj_kernel<<<B_SZ, 256, 0, stream>>>(dec, w2, dp);

    dim3 g2(T_SZ / 128, B_SZ);                  // 8 waves/block * 16 rows/wave
    aa_scores_kernel<<<g2, 256, 0, stream>>>(enc, w1, v, dp, probs);

    aa_softmax_kernel<<<B_SZ, 256, 0, stream>>>(probs);

    aa_context_kernel<<<B_SZ, 256, 0, stream>>>(enc, probs, ctx);
}
